// MultiHeadAttention_10273561772261
// MI455X (gfx1250) — matrix operand order repaired and checked by reading, not compiled, not run on hardware
//
#include <hip/hip_runtime.h>
#include <math.h>

// ---------------------------------------------------------------------------
// MI455X (gfx1250) multi-head causal attention, f16 WMMA path.
//   x:[2,2048,1024] f32, W_q/k/v/o:[1024,1024] f32 (nn.Linear: y = x @ W^T)
// Pipeline:
//   cast f32->f16 -> GEMM(Q*0.125), GEMM(K), GEMM(V^T) -> flash-attention
//   computed TRANSPOSED (S^T = K q^T, O^T = V^T P^T) so each query maps to a
//   lane: softmax state is per-lane scalar, reductions are 1 shuffle each,
//   P^T/O^T move as b128 vectors -> GEMM(out, f32).
// All matmuls: v_wmma_f32_16x16x32_f16, fp32 accumulation.
// ---------------------------------------------------------------------------

typedef _Float16 half_t;
typedef __attribute__((ext_vector_type(16))) _Float16 v16h;
typedef __attribute__((ext_vector_type(8)))  _Float16 v8h;
typedef __attribute__((ext_vector_type(4)))  _Float16 v4h;
typedef __attribute__((ext_vector_type(8)))  float    v8f;
typedef __attribute__((ext_vector_type(4)))  float    v4f;

#define D_MODEL 1024
#define N_HEADS 16
#define D_HEAD  64
#define BATCH   2
#define SEQ     2048
#define M_TOTAL (BATCH * SEQ)   // 4096

// ---------------- fragment helpers (per CDNA5 ISA 7.12.2 layouts) ----------

__device__ __forceinline__ v16h frag_combine(v8h lo, v8h hi) {
  v16h r;
#pragma unroll
  for (int i = 0; i < 8; ++i) { r[i] = lo[i]; r[i + 8] = hi[i]; }
  return r;
}

// A-matrix 16x32 f16: lane holds row M=lane%16; K chunks [h*8, h*8+8) and
// [16+h*8, 16+h*8+8) where h = lane/16.  Source row-major with leading dim lda.
__device__ __forceinline__ v16h load_a_frag(const half_t* __restrict__ A, int lda,
                                            int m0, int k0, int lane) {
  const int r = lane & 15, h = lane >> 4;
  const half_t* p = A + (size_t)(m0 + r) * lda + (k0 + h * 8);
  v8h lo = *(const v8h*)p;
  v8h hi = *(const v8h*)(p + 16);
  return frag_combine(lo, hi);
}

// B-matrix 32x16 f16: lane holds column N=lane%16; K chunks [h*8, h*8+8) and
// [16+h*8, 16+h*8+8).  Source stored N-major: B[n*ldb + k].
__device__ __forceinline__ v16h load_b_frag(const half_t* __restrict__ B, int ldb,
                                            int n0, int k0, int lane) {
  const int r = lane & 15, h = lane >> 4;
  const half_t* p = B + (size_t)(n0 + r) * ldb + (k0 + h * 8);
  v8h lo = *(const v8h*)p;
  v8h hi = *(const v8h*)(p + 16);
  return frag_combine(lo, hi);
}

__device__ __forceinline__ v8f wmma_f16(v16h a, v16h b, v8f c) {
  return __builtin_amdgcn_wmma_f32_16x16x32_f16(false, a, false, b, (short)0, c,
                                                false, false);
}

// ---------------- cast kernel ----------------------------------------------

__global__ void cast_f32_to_f16(const float* __restrict__ src,
                                half_t* __restrict__ dst, int n4) {
  int i = blockIdx.x * blockDim.x + threadIdx.x;
  if (i < n4) {
    v4f f = *(const v4f*)(src + (size_t)i * 4);
    v4h o;
#pragma unroll
    for (int j = 0; j < 4; ++j) o[j] = (half_t)f[j];
    *(v4h*)(dst + (size_t)i * 4) = o;
  }
}

// ---------------- GEMM: y[m,e] = oscale * sum_d A[m,d] * W[e,d] ------------
// One wave computes a 32x32 macro-tile (4 WMMA tiles, operand reuse 2x).
// MODE 0: store f16 head-major [B*H, S, Dh]            (Q, K)
// MODE 2: store f16 head-major transposed [B*H, Dh, S] (V^T)
// MODE 3: store f32 row-major [M, D_MODEL]             (final output)

template <int MODE>
__device__ __forceinline__ void store_tile(void* __restrict__ dst, int m0, int n0,
                                           v8f acc, float oscale, int lane) {
  const int r = lane & 15, h = lane >> 4;
#pragma unroll
  for (int v = 0; v < 8; ++v) {
    const int m = m0 + v + h * 8;        // C/D layout: VGPR v -> row v + h*8
    const int e = n0 + r;
    const float y = acc[v] * oscale;
    if (MODE == 3) {
      ((float*)dst)[(size_t)m * D_MODEL + e] = y;
    } else {
      const int b_ = m / SEQ, s = m - b_ * SEQ;
      const int hh = e / D_HEAD, dh = e - hh * D_HEAD;
      const int bh = b_ * N_HEADS + hh;
      const half_t val = (half_t)y;
      if (MODE == 2)
        ((half_t*)dst)[((size_t)bh * D_HEAD + dh) * SEQ + s] = val;
      else
        ((half_t*)dst)[((size_t)bh * SEQ + s) * D_HEAD + dh] = val;
    }
  }
}

template <int MODE>
__global__ __launch_bounds__(256, 1) void gemm_xwT(const half_t* __restrict__ A,
                                                   const half_t* __restrict__ W,
                                                   void* __restrict__ dst,
                                                   float oscale) {
  const int lane = threadIdx.x & 31;
  const int wave = threadIdx.x >> 5;
  const int tile = blockIdx.x * 8 + wave;
  const int NT = D_MODEL / 32;           // 32 macro-tiles along N
  const int m0 = (tile / NT) * 32;
  const int n0 = (tile % NT) * 32;

  v8f acc00 = {}, acc01 = {}, acc10 = {}, acc11 = {};
#pragma unroll 2
  for (int k0 = 0; k0 < D_MODEL; k0 += 32) {
    v16h a0 = load_a_frag(A, D_MODEL, m0,      k0, lane);
    v16h a1 = load_a_frag(A, D_MODEL, m0 + 16, k0, lane);
    v16h b0 = load_b_frag(W, D_MODEL, n0,      k0, lane);
    v16h b1 = load_b_frag(W, D_MODEL, n0 + 16, k0, lane);
    acc00 = wmma_f16(a0, b0, acc00);
    acc01 = wmma_f16(a0, b1, acc01);
    acc10 = wmma_f16(a1, b0, acc10);
    acc11 = wmma_f16(a1, b1, acc11);
  }
  store_tile<MODE>(dst, m0,      n0,      acc00, oscale, lane);
  store_tile<MODE>(dst, m0,      n0 + 16, acc01, oscale, lane);
  store_tile<MODE>(dst, m0 + 16, n0,      acc10, oscale, lane);
  store_tile<MODE>(dst, m0 + 16, n0 + 16, acc11, oscale, lane);
}

// ---------------- flash attention (causal, transposed) ---------------------
// One wave owns one (batch*head, 16-query) tile; queries live on the LANE
// axis throughout:
//   S^T tile = K_frag(A) x Q_frag(B)    -> D: row=key(v+8h), col=query(r)
//   O^T tile = Vt_frag(A) x P^T_frag(B) -> D: row=dh(v+8h),  col=query(r)
// so rmax/rsum/alpha are per-lane scalars; reductions need one xor-16 shuffle.
// P^T goes through a per-wave LDS tile [query][key] as two b128 stores and is
// read back as a B-fragment (wave-synchronous; per-wave LDS ops are in-order).

template <bool MASK>
__device__ __forceinline__ void attn_chunk(
    int k0, int m0, int lane, int r, int h,
    const half_t* __restrict__ kk, const half_t* __restrict__ vt,
    half_t* PW, const v16h& qb0, const v16h& qb1,
    v8f& acc0, v8f& acc1, v8f& acc2, v8f& acc3,
    float& rmax, float& rsum) {
  // prefetch all 8 A-fragments (4 K, 4 V^T) before any compute
  v16h ka00 = load_a_frag(kk, D_HEAD, k0,      0,  lane);
  v16h ka01 = load_a_frag(kk, D_HEAD, k0,      32, lane);
  v16h ka10 = load_a_frag(kk, D_HEAD, k0 + 16, 0,  lane);
  v16h ka11 = load_a_frag(kk, D_HEAD, k0 + 16, 32, lane);
  v16h va0  = load_a_frag(vt, SEQ, 0,  k0, lane);
  v16h va1  = load_a_frag(vt, SEQ, 16, k0, lane);
  v16h va2  = load_a_frag(vt, SEQ, 32, k0, lane);
  v16h va3  = load_a_frag(vt, SEQ, 48, k0, lane);

  v8f st0 = {}, st1 = {};                // S^T: keys k0..+15 / k0+16..+31
  st0 = wmma_f16(ka00, qb0, st0);
  st0 = wmma_f16(ka01, qb1, st0);
  st1 = wmma_f16(ka10, qb0, st1);
  st1 = wmma_f16(ka11, qb1, st1);

  if (MASK) {
    const int query = m0 + r;
#pragma unroll
    for (int v = 0; v < 8; ++v) {
      const int key = k0 + v + h * 8;
      if (key > query)      st0[v] = -3.0e38f;
      if (key + 16 > query) st1[v] = -3.0e38f;
    }
  }

  // per-query (= per-lane) max over this chunk's 32 keys
  float t = fmaxf(st0[0], st1[0]);
#pragma unroll
  for (int v = 1; v < 8; ++v) t = fmaxf(t, fmaxf(st0[v], st1[v]));
  t = fmaxf(t, __shfl_xor(t, 16));       // combine the two key-halves
  const float nm = fmaxf(rmax, t);
  const float alpha = __expf(rmax - nm);
  rmax = nm;

  v8h p0, p1;
  float sl = 0.0f;
#pragma unroll
  for (int v = 0; v < 8; ++v) {
    const float e0 = __expf(st0[v] - nm);
    const float e1 = __expf(st1[v] - nm);
    sl += e0 + e1;
    p0[v] = (half_t)e0;
    p1[v] = (half_t)e1;
  }
  rsum = rsum * alpha + (sl + __shfl_xor(sl, 16));

  // P^T[query r][keys t*16+8h .. +8): two contiguous b128 LDS stores
  *(v8h*)(PW + r * 32 + 8 * h)       = p0;
  *(v8h*)(PW + r * 32 + 16 + 8 * h)  = p1;

#pragma unroll
  for (int v = 0; v < 8; ++v) {
    acc0[v] *= alpha; acc1[v] *= alpha; acc2[v] *= alpha; acc3[v] *= alpha;
  }

  const v16h pb = load_b_frag(PW, 32, 0, 0, lane);   // P^T as B-fragment
  acc0 = wmma_f16(va0, pb, acc0);
  acc1 = wmma_f16(va1, pb, acc1);
  acc2 = wmma_f16(va2, pb, acc2);
  acc3 = wmma_f16(va3, pb, acc3);
}

__global__ __launch_bounds__(256, 1) void attn_kernel(
    const half_t* __restrict__ Q, const half_t* __restrict__ K,
    const half_t* __restrict__ Vt, half_t* __restrict__ O) {
  __shared__ __attribute__((aligned(16))) half_t Plds[8][16 * 32];

  const int lane = threadIdx.x & 31;
  const int wave = threadIdx.x >> 5;
  const int gw = blockIdx.x * 8 + wave;        // 0 .. 4095
  const int qt = gw & (SEQ / 16 - 1);          // query tile 0..127
  const int bh = gw >> 7;                      // 0..31
  const int m0 = qt * 16;

  const half_t* q  = Q  + (size_t)bh * SEQ * D_HEAD;
  const half_t* kk = K  + (size_t)bh * SEQ * D_HEAD;
  const half_t* vt = Vt + (size_t)bh * D_HEAD * SEQ;
  const int r = lane & 15, h = lane >> 4;
  half_t* PW = &Plds[wave][0];

  // Q as loop-invariant B-fragments (Q was pre-scaled by 1/sqrt(Dh))
  const v16h qb0 = load_b_frag(q, D_HEAD, m0, 0,  lane);
  const v16h qb1 = load_b_frag(q, D_HEAD, m0, 32, lane);

  v8f acc0 = {}, acc1 = {}, acc2 = {}, acc3 = {};   // O^T: 64(dh) x 16(query)
  float rmax = -3.0e38f, rsum = 0.0f;

  // chunks [0, kfull) are strictly below the diagonal; exactly one masked
  // chunk at k0 = kfull covers keys up to m0+15.
  const int kfull = (m0 >= 31) ? ((((m0 - 31) >> 5) << 5) + 32) : 0;
  for (int k0 = 0; k0 < kfull; k0 += 32)
    attn_chunk<false>(k0, m0, lane, r, h, kk, vt, PW, qb0, qb1,
                      acc0, acc1, acc2, acc3, rmax, rsum);
  attn_chunk<true>(kfull, m0, lane, r, h, kk, vt, PW, qb0, qb1,
                   acc0, acc1, acc2, acc3, rmax, rsum);

  // epilogue: O^T D-layout -> row-major O[m, hh*64+dh]; per lane the 8
  // consecutive dh values per tile form one b128 store.
  const float inv = 1.0f / rsum;
  const int b_ = bh >> 4, hh = bh & 15;
  half_t* op = O + ((size_t)(b_ * SEQ + m0 + r)) * D_MODEL + hh * D_HEAD + 8 * h;
  v8h o0, o1, o2, o3;
#pragma unroll
  for (int v = 0; v < 8; ++v) {
    o0[v] = (half_t)(acc0[v] * inv);
    o1[v] = (half_t)(acc1[v] * inv);
    o2[v] = (half_t)(acc2[v] * inv);
    o3[v] = (half_t)(acc3[v] * inv);
  }
  *(v8h*)(op + 0)  = o0;
  *(v8h*)(op + 16) = o1;
  *(v8h*)(op + 32) = o2;
  *(v8h*)(op + 48) = o3;
}

// ---------------- host launcher --------------------------------------------

extern "C" void kernel_launch(void* const* d_in, const int* in_sizes, int n_in,
                              void* d_out, int out_size, void* d_ws, size_t ws_size,
                              hipStream_t stream) {
  const float* x  = (const float*)d_in[0];
  const float* Wq = (const float*)d_in[1];
  const float* Wk = (const float*)d_in[2];
  const float* Wv = (const float*)d_in[3];
  const float* Wo = (const float*)d_in[4];

  char* ws = (char*)d_ws;
  size_t off = 0;
  auto alloc = [&](size_t bytes) -> char* {
    char* p = ws + off;
    off += (bytes + 255) & ~(size_t)255;
    return p;
  };
  // ~48 MiB of workspace (harness-provided d_ws).
  half_t* xb  = (half_t*)alloc((size_t)M_TOTAL * D_MODEL * sizeof(half_t));
  half_t* Wqb = (half_t*)alloc((size_t)D_MODEL * D_MODEL * sizeof(half_t));
  half_t* Wkb = (half_t*)alloc((size_t)D_MODEL * D_MODEL * sizeof(half_t));
  half_t* Wvb = (half_t*)alloc((size_t)D_MODEL * D_MODEL * sizeof(half_t));
  half_t* Wob = (half_t*)alloc((size_t)D_MODEL * D_MODEL * sizeof(half_t));
  half_t* Qb  = (half_t*)alloc((size_t)M_TOTAL * D_MODEL * sizeof(half_t));
  half_t* Kb  = (half_t*)alloc((size_t)M_TOTAL * D_MODEL * sizeof(half_t));
  half_t* Vtb = (half_t*)alloc((size_t)M_TOTAL * D_MODEL * sizeof(half_t));
  half_t* Ob  = (half_t*)alloc((size_t)M_TOTAL * D_MODEL * sizeof(half_t));

  const int nx4 = (M_TOTAL * D_MODEL) / 4;   // 1,048,576
  const int nw4 = (D_MODEL * D_MODEL) / 4;   //   262,144
  cast_f32_to_f16<<<nx4 / 256, 256, 0, stream>>>(x,  xb,  nx4);
  cast_f32_to_f16<<<nw4 / 256, 256, 0, stream>>>(Wq, Wqb, nw4);
  cast_f32_to_f16<<<nw4 / 256, 256, 0, stream>>>(Wk, Wkb, nw4);
  cast_f32_to_f16<<<nw4 / 256, 256, 0, stream>>>(Wv, Wvb, nw4);
  cast_f32_to_f16<<<nw4 / 256, 256, 0, stream>>>(Wo, Wob, nw4);

  // (4096/32)*(1024/32) = 4096 macro-tiles / 8 waves-per-block = 512 blocks
  const int gemm_blocks = (M_TOTAL / 32) * (D_MODEL / 32) / 8;
  gemm_xwT<0><<<gemm_blocks, 256, 0, stream>>>(xb, Wqb, (void*)Qb, 0.125f);
  gemm_xwT<0><<<gemm_blocks, 256, 0, stream>>>(xb, Wkb, (void*)Kb, 1.0f);
  gemm_xwT<2><<<gemm_blocks, 256, 0, stream>>>(xb, Wvb, (void*)Vtb, 1.0f);

  // 32 (b,h) * 128 query tiles = 4096 waves / 8 per block = 512 blocks
  attn_kernel<<<(BATCH * N_HEADS) * (SEQ / 16) / 8, 256, 0, stream>>>(Qb, Kb, Vtb, Ob);

  gemm_xwT<3><<<gemm_blocks, 256, 0, stream>>>(Ob, Wob, d_out, 1.0f);
}
